// MultiHeadSelfAttention_35055523070395
// MI455X (gfx1250) — matrix operand order repaired and checked by reading, not compiled, not run on hardware
//
#include <hip/hip_runtime.h>
#include <hip/hip_bf16.h>
#include <stdint.h>

// Problem constants (from the reference)
constexpr int Bv  = 2;
constexpr int Sv  = 2048;
constexpr int Dv  = 1024;
constexpr int Hv  = 16;
constexpr int DKv = 64;
constexpr int Mv  = Bv * Sv;          // 4096 rows for the projection GEMMs

// ---------- WMMA types ----------
typedef __attribute__((ext_vector_type(16))) __bf16        v16bf;
typedef __attribute__((ext_vector_type(8)))  float         v8f;
typedef __attribute__((ext_vector_type(4)))  unsigned int  v4u;

union FragBF {
  v16bf          v;
  v4u            q[2];
  unsigned int   w[8];
  unsigned short u[16];
};

__device__ __forceinline__ v8f wmma_bf16(const FragBF& a, const FragBF& b, v8f c) {
  // emits v_wmma_f32_16x16x32_bf16
  return __builtin_amdgcn_wmma_f32_16x16x32_bf16(
      false, a.v, false, b.v, (short)0, c, false, false);
}

__device__ __forceinline__ unsigned short f32_to_bf16(float f) {
  unsigned int u = __builtin_bit_cast(unsigned int, f);
  unsigned int lsb = (u >> 16) & 1u;
  u += 0x7fffu + lsb;                 // round-to-nearest-even
  return (unsigned short)(u >> 16);
}

// ---------- f32 -> bf16 (straight) ----------
__global__ void cvt_f32_bf16(const float* __restrict__ in,
                             unsigned short* __restrict__ out, int n) {
  int i = blockIdx.x * blockDim.x + threadIdx.x;
  if (i < n) out[i] = f32_to_bf16(in[i]);
}

// ---------- f32 [K,N] -> bf16 transposed [N,K] (for WMMA B operands) ----------
__global__ void cvt_transpose_bf16(const float* __restrict__ in,
                                   unsigned short* __restrict__ out,
                                   int K, int N) {
  int id = blockIdx.x * blockDim.x + threadIdx.x;   // id = n*K + k
  if (id >= K * N) return;
  int n = id / K, k = id - n * K;
  out[id] = f32_to_bf16(in[k * N + n]);
}

// ---------- V: f32 [B,S,D] -> bf16 [B,D,S] (so PV B-fragments are contiguous) ----------
__global__ void transpose_v_bf16(const float* __restrict__ in,
                                 unsigned short* __restrict__ out) {
  int id = blockIdx.x * blockDim.x + threadIdx.x;   // id = (b*D + d)*S + s
  if (id >= Bv * Dv * Sv) return;
  int s  = id & (Sv - 1);
  int d  = (id >> 11) & (Dv - 1);                   // Sv = 2048 = 2^11
  int b  = id >> 22;                                // Dv*Sv = 2^21... (2*2^21) -> b in {0,1}
  out[id] = f32_to_bf16(in[(b * Sv + s) * Dv + d]);
}

// ---------- RoPE over full D (reference applies rope before head split) ----------
__global__ void rope_f32_to_bf16(const float* __restrict__ in,
                                 unsigned short* __restrict__ out) {
  int idx = blockIdx.x * blockDim.x + threadIdx.x;  // pair index
  const int total = Bv * Sv * (Dv / 2);
  if (idx >= total) return;
  int p  = idx % (Dv / 2);
  int bs = idx / (Dv / 2);
  int s  = bs % Sv;
  float inv = __expf((float)(-2 * p) * (9.210340371976184f / (float)Dv));
  float ang = (float)s * inv;
  float sn, cs;
  __sincosf(ang, &sn, &cs);
  const int base = bs * Dv + 2 * p;
  float x1 = in[base], x2 = in[base + 1];
  out[base]     = f32_to_bf16(x1 * cs - x2 * sn);
  out[base + 1] = f32_to_bf16(x1 * sn + x2 * cs);
}

// ---------- async copy of a 64x32 bf16 A tile into LDS (256 x 16B segments) ----------
__device__ __forceinline__ void async_copy_tileA(
    const unsigned short* __restrict__ A, int K, int m0, int k0,
    unsigned lbase, int tid) {
#pragma unroll
  for (int j = 0; j < 2; ++j) {
    const int id  = tid + 128 * j;                  // 0..255
    const int row = id >> 2, seg = id & 3;
    const unsigned short* g = A + (m0 + row) * K + k0 + seg * 8;
    const unsigned l = lbase + (unsigned)(row * 64 + seg * 16);
    asm volatile("global_load_async_to_lds_b128 %0, %1, off"
                 :: "v"(l), "v"(g) : "memory");
  }
}

// ---------- bf16 WMMA GEMM: C[M,N] = A[M,K] * Wt[N,K]^T, f32 accumulate ----------
// Block = 4 waves sharing a 64-row async-staged A tile; wave tile = 64x32 (8 WMMA/k).
__global__ __launch_bounds__(128) void gemm_bf16_wmma(
    const unsigned short* __restrict__ A,    // [M,K] bf16 row-major
    const unsigned short* __restrict__ Wt,   // [N,K] bf16 (transposed weight)
    float* __restrict__ C,                   // [M,N] f32
    int M, int N, int K) {
  __shared__ unsigned short bufA[2][64 * 32];      // 2 x 4KB double buffer
  const int tid  = threadIdx.x;
  const int wave = tid >> 5, lane = tid & 31;
  const int hh   = lane >> 4, r = lane & 15;
  const int tilesN = N / 128;
  const int m0 = (blockIdx.x / tilesN) * 64;
  const int n0 = (blockIdx.x % tilesN) * 128 + wave * 32;

  const unsigned lb0 = (unsigned)(uintptr_t)&bufA[0][0];
  const unsigned lb1 = (unsigned)(uintptr_t)&bufA[1][0];

  v8f acc[4][2];
#pragma unroll
  for (int f = 0; f < 4; ++f) { acc[f][0] = {}; acc[f][1] = {}; }

  const unsigned short* bcol0 = Wt + (n0 + r) * K + hh * 16;        // B col n0+r
  const unsigned short* bcol1 = Wt + (n0 + 16 + r) * K + hh * 16;   // B col n0+16+r

  async_copy_tileA(A, K, m0, 0, lb0, tid);          // prologue: tile 0

  for (int k0 = 0; k0 < K; k0 += 32) {
    const int cur = (k0 >> 5) & 1;
    __syncthreads();                                // prev compute done in all waves
    const bool more = (k0 + 32 < K);
    if (more) {
      async_copy_tileA(A, K, m0, k0 + 32, cur ? lb0 : lb1, tid);
      asm volatile("s_wait_asynccnt 0x2" ::: "memory");   // current tile landed
    } else {
      asm volatile("s_wait_asynccnt 0x0" ::: "memory");
    }
    __syncthreads();                                // tile visible to all waves

    const unsigned short* lbuf = &bufA[cur][0];
    FragBF a[4], b0, b1;
#pragma unroll
    for (int f = 0; f < 4; ++f) {
      const int row = f * 16 + r;
      a[f].q[0] = *(const v4u*)&lbuf[row * 32 + hh * 16];       // ds_load_b128
      a[f].q[1] = *(const v4u*)&lbuf[row * 32 + hh * 16 + 8];
    }
    if (more) __builtin_prefetch(bcol0 + k0 + 32, 0, 1);        // global_prefetch_b8
    b0.q[0] = *(const v4u*)(bcol0 + k0);                        // global_load_b128
    b0.q[1] = *(const v4u*)(bcol0 + k0 + 8);
    b1.q[0] = *(const v4u*)(bcol1 + k0);
    b1.q[1] = *(const v4u*)(bcol1 + k0 + 8);
#pragma unroll
    for (int f = 0; f < 4; ++f) {
      acc[f][0] = wmma_bf16(a[f], b0, acc[f][0]);
      acc[f][1] = wmma_bf16(a[f], b1, acc[f][1]);
    }
  }
  // C layout: VGPR j -> row hh*8+j (within 16-row tile), col = base + r
#pragma unroll
  for (int f = 0; f < 4; ++f) {
#pragma unroll
    for (int j = 0; j < 8; ++j) {
      const int row = m0 + f * 16 + hh * 8 + j;
      C[row * N + n0 + r]      = acc[f][0][j];
      C[row * N + n0 + 16 + r] = acc[f][1][j];
    }
  }
}

// ---------- Causal flash attention (bf16 WMMA, f32 online softmax) ----------
// Grid: x = S/16 query tiles, y = (B*H)/4; wave = one (b,h) head, 16 queries.
// All 4 waves in a block share q0 -> uniform loop count -> barriers are safe.
__global__ __launch_bounds__(128) void attn_fwd_wmma(
    const unsigned short* __restrict__ Q,   // [B,S,D] bf16 (rope'd)
    const unsigned short* __restrict__ K,   // [B,S,D] bf16 (rope'd)
    const unsigned short* __restrict__ Vt,  // [B,D,S] bf16 (transposed)
    unsigned short* __restrict__ O) {       // [B,S,D] bf16
  __shared__ unsigned short pbuf[4][16 * 32];     // per-wave bf16 P staging (4KB)

  const int wave = threadIdx.x >> 5;
  const int lane = threadIdx.x & 31;
  const int hh   = lane >> 4;
  const int r    = lane & 15;
  const int q0   = blockIdx.x * 16;
  const int bh   = blockIdx.y * 4 + wave;
  const int b    = bh >> 4;                 // / H
  const int h    = bh & 15;                 // % H
  const int headBase = b * Sv * Dv + h * DKv;

  // Q A-fragments for dk 0..31 and 32..63 (loaded once, 4 x b128)
  FragBF qa0, qa1;
  {
    const unsigned short* qrow = Q + headBase + (q0 + r) * Dv + hh * 16;
    qa0.q[0] = *(const v4u*)(qrow);
    qa0.q[1] = *(const v4u*)(qrow + 8);
    qa1.q[0] = *(const v4u*)(qrow + 32);
    qa1.q[1] = *(const v4u*)(qrow + 40);
  }

  float mrow[8], lrow[8];
#pragma unroll
  for (int j = 0; j < 8; ++j) { mrow[j] = -1e30f; lrow[j] = 0.0f; }
  v8f oacc[4] = {{}, {}, {}, {}};           // 16 x 64 output accumulator

  const int ntiles = (q0 + 16 + 31) / 32;   // causal: keys up to q0+15
  for (int kt = 0; kt < ntiles; ++kt) {
    const int kb = kt * 32;
    // ---- scores: S = Q (16x64) @ K^T (64x32) as two 16x16 tiles ----
    v8f s0 = {}, s1 = {};
    {
      FragBF kf;
      const unsigned short* krow0 = K + headBase + (kb + r) * Dv + hh * 16;
      kf.q[0] = *(const v4u*)(krow0);
      kf.q[1] = *(const v4u*)(krow0 + 8);
      s0 = wmma_bf16(qa0, kf, s0);
      kf.q[0] = *(const v4u*)(krow0 + 32);
      kf.q[1] = *(const v4u*)(krow0 + 40);
      s0 = wmma_bf16(qa1, kf, s0);
      const unsigned short* krow1 = krow0 + 16 * Dv;
      kf.q[0] = *(const v4u*)(krow1);
      kf.q[1] = *(const v4u*)(krow1 + 8);
      s1 = wmma_bf16(qa0, kf, s1);
      kf.q[0] = *(const v4u*)(krow1 + 32);
      kf.q[1] = *(const v4u*)(krow1 + 40);
      s1 = wmma_bf16(qa1, kf, s1);
    }
    // ---- online softmax: scale, causal mask, half-wave row max/sum ----
    const int col0 = kb + r, col1 = kb + 16 + r;
    float p0[8], p1[8];
#pragma unroll
    for (int j = 0; j < 8; ++j) {
      const int row = q0 + hh * 8 + j;
      float v0 = s0[j] * 0.125f;            // 1/sqrt(64)
      float v1 = s1[j] * 0.125f;
      if (col0 > row) v0 = -1e30f;
      if (col1 > row) v1 = -1e30f;
      float mx = fmaxf(v0, v1);
      mx = fmaxf(mx, __shfl_xor(mx, 1));
      mx = fmaxf(mx, __shfl_xor(mx, 2));
      mx = fmaxf(mx, __shfl_xor(mx, 4));
      mx = fmaxf(mx, __shfl_xor(mx, 8));    // stays within the 16-lane half
      const float mnew  = fmaxf(mrow[j], mx);
      const float alpha = __expf(mrow[j] - mnew);
      p0[j] = __expf(v0 - mnew);
      p1[j] = __expf(v1 - mnew);
      float sm = p0[j] + p1[j];
      sm += __shfl_xor(sm, 1);
      sm += __shfl_xor(sm, 2);
      sm += __shfl_xor(sm, 4);
      sm += __shfl_xor(sm, 8);
      lrow[j] = lrow[j] * alpha + sm;
      mrow[j] = mnew;
      oacc[0][j] *= alpha; oacc[1][j] *= alpha;
      oacc[2][j] *= alpha; oacc[3][j] *= alpha;
    }
    // ---- transpose P (C-layout) to A-fragment layout via LDS, bf16 ----
#pragma unroll
    for (int j = 0; j < 8; ++j) {
      pbuf[wave][(hh * 8 + j) * 32 + r]      = f32_to_bf16(p0[j]);
      pbuf[wave][(hh * 8 + j) * 32 + 16 + r] = f32_to_bf16(p1[j]);
    }
    __syncthreads();
    FragBF pa;
    pa.q[0] = *(const v4u*)&pbuf[wave][r * 32 + hh * 16];       // ds_load_b128
    pa.q[1] = *(const v4u*)&pbuf[wave][r * 32 + hh * 16 + 8];
    __syncthreads();
    // ---- O += P (16x32) @ V (32x64): four 16-wide N tiles from Vt ----
#pragma unroll
    for (int t = 0; t < 4; ++t) {
      FragBF vf;
      const unsigned short* vrow =
          Vt + (b * Dv + h * DKv + t * 16 + r) * Sv + kb + hh * 16;
      vf.q[0] = *(const v4u*)(vrow);
      vf.q[1] = *(const v4u*)(vrow + 8);
      oacc[t] = wmma_bf16(pa, vf, oacc[t]);
    }
  }
  // ---- normalize and store bf16 ----
#pragma unroll
  for (int j = 0; j < 8; ++j) {
    const float inv = 1.0f / lrow[j];
    const int orow = headBase + (q0 + hh * 8 + j) * Dv;
#pragma unroll
    for (int t = 0; t < 4; ++t)
      O[orow + t * 16 + r] = f32_to_bf16(oacc[t][j] * inv);
  }
}

// ---------- host-side orchestration ----------
extern "C" void kernel_launch(void* const* d_in, const int* in_sizes, int n_in,
                              void* d_out, int out_size, void* d_ws, size_t ws_size,
                              hipStream_t stream) {
  const float* x  = (const float*)d_in[0];
  const float* Wq = (const float*)d_in[1];
  const float* Wk = (const float*)d_in[2];
  const float* Wv = (const float*)d_in[3];
  const float* Wo = (const float*)d_in[4];
  float* out = (float*)d_out;

  // Workspace layout (88 MiB needed)
  char* ws = (char*)d_ws;
  const size_t MiB = (size_t)1 << 20;
  unsigned short* x_bf  = (unsigned short*)(ws + 0);          // 8 MiB
  unsigned short* wq_t  = (unsigned short*)(ws + 8  * MiB);   // 2 MiB each, [N,K]
  unsigned short* wk_t  = (unsigned short*)(ws + 10 * MiB);
  unsigned short* wv_t  = (unsigned short*)(ws + 12 * MiB);
  unsigned short* wo_t  = (unsigned short*)(ws + 14 * MiB);
  float* Qf = (float*)(ws + 16 * MiB);                        // 16 MiB each
  float* Kf = (float*)(ws + 32 * MiB);
  float* Vf = (float*)(ws + 48 * MiB);
  unsigned short* Qbf = (unsigned short*)(ws + 64 * MiB);     // 8 MiB each
  unsigned short* Kbf = (unsigned short*)(ws + 72 * MiB);
  unsigned short* Vt  = (unsigned short*)(ws + 80 * MiB);     // [B,D,S]
  unsigned short* attn_bf = (unsigned short*)(ws + 16 * MiB); // alias Qf (dead after rope)

  const int NX = Bv * Sv * Dv;  // 4,194,304
  const int NW = Dv * Dv;       // 1,048,576

  // 1) narrow x; transpose+narrow weights (WMMA B operands want [N,K])
  cvt_f32_bf16<<<(NX + 255) / 256, 256, 0, stream>>>(x, x_bf, NX);
  cvt_transpose_bf16<<<(NW + 255) / 256, 256, 0, stream>>>(Wq, wq_t, Dv, Dv);
  cvt_transpose_bf16<<<(NW + 255) / 256, 256, 0, stream>>>(Wk, wk_t, Dv, Dv);
  cvt_transpose_bf16<<<(NW + 255) / 256, 256, 0, stream>>>(Wv, wv_t, Dv, Dv);
  cvt_transpose_bf16<<<(NW + 255) / 256, 256, 0, stream>>>(Wo, wo_t, Dv, Dv);

  // 2) QKV projections: [4096,1024] x [1024,1024]
  const int gemmBlocks = (Mv / 64) * (Dv / 128);   // 64 * 8 = 512
  gemm_bf16_wmma<<<gemmBlocks, 128, 0, stream>>>(x_bf, wq_t, Qf, Mv, Dv, Dv);
  gemm_bf16_wmma<<<gemmBlocks, 128, 0, stream>>>(x_bf, wk_t, Kf, Mv, Dv, Dv);
  gemm_bf16_wmma<<<gemmBlocks, 128, 0, stream>>>(x_bf, wv_t, Vf, Mv, Dv, Dv);

  // 3) RoPE on Q,K (over full D, per reference); V -> bf16 [B,D,S]
  const int NP = NX / 2;
  rope_f32_to_bf16<<<(NP + 255) / 256, 256, 0, stream>>>(Qf, Qbf);
  rope_f32_to_bf16<<<(NP + 255) / 256, 256, 0, stream>>>(Kf, Kbf);
  transpose_v_bf16<<<(NX + 255) / 256, 256, 0, stream>>>(Vf, Vt);

  // 4) causal flash attention
  dim3 agrid(Sv / 16, (Bv * Hv) / 4);
  attn_fwd_wmma<<<agrid, 128, 0, stream>>>(Qbf, Kbf, Vt, attn_bf);

  // 5) output projection -> f32 d_out
  gemm_bf16_wmma<<<gemmBlocks, 128, 0, stream>>>(attn_bf, wo_t, out, Mv, Dv, Dv);
}
